// BinSpecCNN_609885356865
// MI455X (gfx1250) — 8-bit matrix operand order repaired and checked by reading, compiled, not run on hardware
//
#include <hip/hip_runtime.h>
#include <stdint.h>

typedef __attribute__((ext_vector_type(8))) int v8i;

#define BN_EPS 1e-5f
#define KPAD_MAX 832
#define ATILE_MAX 3456

__global__ void bn_stats_kernel(const float* __restrict__ x, float* __restrict__ stats,
                                int C, int N, int HW) {
  int c = blockIdx.x;
  int tid = threadIdx.x;
  long cnt = (long)N * HW;
  float s = 0.f, s2 = 0.f;
  for (long i = tid; i < cnt; i += blockDim.x) {
    long n = i / HW;
    long hw = i - n * HW;
    float v = x[((long)n * C + c) * HW + hw];
    s += v;
    s2 += v * v;
  }
  __shared__ float ls[256];
  __shared__ float lq[256];
  ls[tid] = s; lq[tid] = s2;
  __syncthreads();
  for (int off = 128; off > 0; off >>= 1) {
    if (tid < off) { ls[tid] += ls[tid + off]; lq[tid] += lq[tid + off]; }
    __syncthreads();
  }
  if (tid == 0) {
    float mean = ls[0] / (float)cnt;
    float var = lq[0] / (float)cnt - mean * mean;
    stats[2 * c] = mean;
    stats[2 * c + 1] = rsqrtf(var + BN_EPS);
  }
}

__global__ void bn_bin_kernel(const float* __restrict__ x, const float* __restrict__ stats,
                              const float* __restrict__ g, const float* __restrict__ b,
                              int8_t* __restrict__ out, int C, int HW, long total) {
  long i = (long)blockIdx.x * blockDim.x + threadIdx.x;
  if (i >= total) return;
  int c = (int)((i / HW) % C);
  float v = (x[i] - stats[2 * c]) * stats[2 * c + 1] * g[c] + b[c];
  out[i] = v > 0.f ? (int8_t)1 : (v < 0.f ? (int8_t)-1 : (int8_t)0);
}

__global__ void bin_w_kernel(const float* __restrict__ w, int8_t* __restrict__ out, int n) {
  int i = blockIdx.x * blockDim.x + threadIdx.x;
  if (i >= n) return;
  float v = w[i];
  out[i] = v > 0.f ? (int8_t)1 : (v < 0.f ? (int8_t)-1 : (int8_t)0);
}

__global__ void maxpool_kernel(const float* __restrict__ in, float* __restrict__ out,
                               int NC, int H, int W, int OH, int OW, int s) {
  long i = (long)blockIdx.x * blockDim.x + threadIdx.x;
  long total = (long)NC * OH * OW;
  if (i >= total) return;
  int ox = (int)(i % OW);
  long t = i / OW;
  int oy = (int)(t % OH);
  int nc = (int)(t / OH);
  const float* p = in + ((long)nc * H + (long)oy * s) * W + (long)ox * s;
  float m = -INFINITY;
#pragma unroll
  for (int dy = 0; dy < 3; ++dy)
#pragma unroll
    for (int dx = 0; dx < 3; ++dx) {
      float v = p[(long)dy * W + dx];
      m = v > m ? v : m;
    }
  out[i] = m;
}

__device__ __forceinline__ void load_ab64(const int8_t* __restrict__ arow,
                                          const int8_t* __restrict__ wrow,
                                          int k0, int h, v8i& A, v8i& B) {
#pragma unroll
  for (int r = 0; r < 8; ++r) {
    int ka = k0 + ((r >> 2) << 5) + (((r >> 1) & 1) << 4) + ((r & 1) << 2) + (h << 3);
    A[r] = *(const int*)(arow + ka);
    int kb = k0 + ((r >> 2) << 5) + (((r >> 1) & 1) << 4) + ((r & 1) << 2) + (h << 3);
    B[r] = *(const int*)(wrow + kb);
  }
}

template <int KH, int KW>
__global__ void __launch_bounds__(32)
binconv_wmma_kernel(const int8_t* __restrict__ act, const int8_t* __restrict__ wts,
                    float* __restrict__ out,
                    int Cin, int Cout, int H, int W, int OH, int OW, int ocTiles) {
  __shared__ __align__(16) int8_t wtile[16 * KPAD_MAX];
  __shared__ __align__(16) int8_t a2[16 * KPAD_MAX];
  __shared__ __align__(16) int8_t atile[ATILE_MAX + 16];

  const int lane = threadIdx.x;
  const int oxbase = blockIdx.x * 16;
  const int oy = blockIdx.y;
  const int n = blockIdx.z / ocTiles;
  const int octile = blockIdx.z % ocTiles;

  const int Keff = Cin * KH * KW;
  const int Kpad = (Keff + 63) & ~63;
  const int IW = KW + 15;
  const int Apatch = Cin * KH * IW;
  const int ZOFF = Apatch;

  __builtin_prefetch(&wts[(long)(octile * 16) * Keff], 0, 1);

  for (int idx = lane; idx < 16 * Kpad; idx += 32) {
    int oc = idx / Kpad;
    int k = idx - oc * Kpad;
    int goc = octile * 16 + oc;
    int8_t val = 0;
    if (goc < Cout && k < Keff) val = wts[(long)goc * Keff + k];
    wtile[oc * Kpad + k] = val;
  }
  for (int idx = lane; idx < Apatch; idx += 32) {
    int c = idx / (KH * IW);
    int r = idx - c * (KH * IW);
    int ky = r / IW;
    int ix = oxbase + (r - ky * IW);
    int iy = oy + ky;
    int8_t val = 0;
    if (ix < W) val = act[(((long)n * Cin + c) * H + iy) * W + ix];
    atile[idx] = val;
  }
  if (lane < 16) atile[ZOFF + lane] = 0;
  __syncthreads();

  for (int k = lane; k < Kpad; k += 32) {
    int off = ZOFF;
    if (k < Keff) {
      int c = k / (KH * KW);
      int r = k - c * (KH * KW);
      int ky = r / KW;
      int kx = r - ky * KW;
      off = (c * KH + ky) * IW + kx;
    }
#pragma unroll
    for (int mm = 0; mm < 16; ++mm) a2[mm * Kpad + k] = atile[off + mm];
  }
  __syncthreads();

  const int m = lane & 15;
  const int h = lane >> 4;
  const int8_t* arow = &a2[m * Kpad];
  const int8_t* wrow = &wtile[m * Kpad];

  v8i acc0 = {0, 0, 0, 0, 0, 0, 0, 0};
  v8i acc1 = {0, 0, 0, 0, 0, 0, 0, 0};
  const int nchunks = Kpad >> 6;
  int kc = 0;
  for (; kc + 1 < nchunks; kc += 2) {
    v8i A0, B0, A1, B1;
    load_ab64(arow, wrow, kc << 6, h, A0, B0);
    load_ab64(arow, wrow, (kc + 1) << 6, h, A1, B1);
    acc0 = __builtin_amdgcn_wmma_i32_16x16x64_iu8(true, A0, true, B0, acc0, false, false);
    acc1 = __builtin_amdgcn_wmma_i32_16x16x64_iu8(true, A1, true, B1, acc1, false, false);
  }
  if (kc < nchunks) {
    v8i A0, B0;
    load_ab64(arow, wrow, kc << 6, h, A0, B0);
    acc0 = __builtin_amdgcn_wmma_i32_16x16x64_iu8(true, A0, true, B0, acc0, false, false);
  }
  v8i acc = acc0 + acc1;

  const int oc = octile * 16 + m;
  if (oc < Cout) {
#pragma unroll
    for (int i = 0; i < 8; ++i) {
      int ox = oxbase + i + 8 * h;
      if (ox < OW)
        out[(((long)n * Cout + oc) * OH + oy) * OW + ox] = (float)acc[i];
    }
  }
}

__global__ void __launch_bounds__(32)
fc_wmma_kernel(const int8_t* __restrict__ act, const int8_t* __restrict__ wts,
               float* __restrict__ out, int Ncls, int K) {
  const int lane = threadIdx.x;
  const int m = lane & 15;
  const int h = lane >> 4;
  const int row = blockIdx.x * 16 + m;
  const int col = blockIdx.y * 16 + m;
  const int colc = col < Ncls ? col : (Ncls - 1);
  const int8_t* arow = &act[(long)row * K];
  const int8_t* wrow = &wts[(long)colc * K];

  v8i acc0 = {0, 0, 0, 0, 0, 0, 0, 0};
  v8i acc1 = {0, 0, 0, 0, 0, 0, 0, 0};
  int k0 = 0;
  for (; k0 + 128 <= K; k0 += 128) {
    v8i A0, B0, A1, B1;
    load_ab64(arow, wrow, k0, h, A0, B0);
    load_ab64(arow, wrow, k0 + 64, h, A1, B1);
    acc0 = __builtin_amdgcn_wmma_i32_16x16x64_iu8(true, A0, true, B0, acc0, false, false);
    acc1 = __builtin_amdgcn_wmma_i32_16x16x64_iu8(true, A1, true, B1, acc1, false, false);
  }
  for (; k0 < K; k0 += 64) {
    v8i A0, B0;
    load_ab64(arow, wrow, k0, h, A0, B0);
    acc0 = __builtin_amdgcn_wmma_i32_16x16x64_iu8(true, A0, true, B0, acc0, false, false);
  }
  v8i acc = acc0 + acc1;

  if (col < Ncls) {
#pragma unroll
    for (int i = 0; i < 8; ++i) {
      int rr = blockIdx.x * 16 + i + 8 * h;
      out[(long)rr * Ncls + col] = (float)acc[i];
    }
  }
}

static inline size_t alignup256(size_t v) { return (v + 255) & ~(size_t)255; }

extern "C" void kernel_launch(void* const* d_in, const int* in_sizes, int n_in,
                              void* d_out, int out_size, void* d_ws, size_t ws_size,
                              hipStream_t stream) {
  (void)in_sizes; (void)n_in; (void)out_size; (void)ws_size;
  const float* x = (const float*)d_in[0];
  const float* gs[7] = {(const float*)d_in[1], (const float*)d_in[4], (const float*)d_in[7],
                        (const float*)d_in[10], (const float*)d_in[13], (const float*)d_in[16],
                        (const float*)d_in[19]};
  const float* bs[7] = {(const float*)d_in[2], (const float*)d_in[5], (const float*)d_in[8],
                        (const float*)d_in[11], (const float*)d_in[14], (const float*)d_in[17],
                        (const float*)d_in[20]};
  const float* wf[7] = {(const float*)d_in[3], (const float*)d_in[6], (const float*)d_in[9],
                        (const float*)d_in[12], (const float*)d_in[15], (const float*)d_in[18],
                        (const float*)d_in[21]};
  const int wsz[7] = {8 * 3 * 11 * 11, 16 * 8 * 7 * 7, 32 * 16 * 5 * 5,
                      32 * 32 * 5 * 5, 64 * 32 * 5 * 5, 64 * 64 * 3 * 3, 50 * 256};

  char* ws = (char*)d_ws;
  size_t o = 0;
  float* convbuf = (float*)(ws + o); o += alignup256((size_t)43454464 * 4);
  float* poolbuf = (float*)(ws + o); o += alignup256((size_t)23011328 * 4);
  int8_t* actbuf = (int8_t*)(ws + o); o += alignup256((size_t)23011328);
  int8_t* wb[7];
  for (int i = 0; i < 7; ++i) { wb[i] = (int8_t*)(ws + o); o += alignup256((size_t)wsz[i]); }
  float* stats = (float*)(ws + o); o += alignup256(128 * sizeof(float));

  for (int i = 0; i < 7; ++i)
    bin_w_kernel<<<(wsz[i] + 255) / 256, 256, 0, stream>>>(wf[i], wb[i], wsz[i]);

  const int N = 64;
  auto bn_bin = [&](const float* src, int C, int HW, int layer) {
    bn_stats_kernel<<<C, 256, 0, stream>>>(src, stats, C, N, HW);
    long total = (long)N * C * HW;
    bn_bin_kernel<<<(int)((total + 255) / 256), 256, 0, stream>>>(
        src, stats, gs[layer], bs[layer], actbuf, C, HW, total);
  };
  auto pool = [&](int NC, int H, int W, int OH, int OW, int s) {
    long total = (long)NC * OH * OW;
    maxpool_kernel<<<(int)((total + 255) / 256), 256, 0, stream>>>(
        convbuf, poolbuf, NC, H, W, OH, OW, s);
  };

  bn_bin(x, 3, 224 * 224, 0);
  binconv_wmma_kernel<11, 11><<<dim3(14, 214, 64 * 1), 32, 0, stream>>>(
      actbuf, wb[0], convbuf, 3, 8, 224, 224, 214, 214, 1);
  pool(64 * 8, 214, 214, 212, 212, 1);

  bn_bin(poolbuf, 8, 212 * 212, 1);
  binconv_wmma_kernel<7, 7><<<dim3(13, 206, 64 * 1), 32, 0, stream>>>(
      actbuf, wb[1], convbuf, 8, 16, 212, 212, 206, 206, 1);
  pool(64 * 16, 206, 206, 102, 102, 2);

  bn_bin(poolbuf, 16, 102 * 102, 2);
  binconv_wmma_kernel<5, 5><<<dim3(7, 98, 64 * 2), 32, 0, stream>>>(
      actbuf, wb[2], convbuf, 16, 32, 102, 102, 98, 98, 2);
  pool(64 * 32, 98, 98, 48, 48, 2);

  bn_bin(poolbuf, 32, 48 * 48, 3);
  binconv_wmma_kernel<5, 5><<<dim3(3, 44, 64 * 2), 32, 0, stream>>>(
      actbuf, wb[3], convbuf, 32, 32, 48, 48, 44, 44, 2);
  pool(64 * 32, 44, 44, 21, 21, 2);

  bn_bin(poolbuf, 32, 21 * 21, 4);
  binconv_wmma_kernel<5, 5><<<dim3(2, 17, 64 * 4), 32, 0, stream>>>(
      actbuf, wb[4], convbuf, 32, 64, 21, 21, 17, 17, 4);
  pool(64 * 64, 17, 17, 8, 8, 2);

  bn_bin(poolbuf, 64, 8 * 8, 5);
  binconv_wmma_kernel<3, 3><<<dim3(1, 6, 64 * 4), 32, 0, stream>>>(
      actbuf, wb[5], convbuf, 64, 64, 8, 8, 6, 6, 4);
  pool(64 * 64, 6, 6, 2, 2, 2);

  bn_bin(poolbuf, 64, 2 * 2, 6);
  fc_wmma_kernel<<<dim3(4, 4), 32, 0, stream>>>(actbuf, wb[6], (float*)d_out, 50, 256);
}
